// SelfAttentionHybrid_65481071406728
// MI455X (gfx1250) — matrix operand order repaired and checked by reading, not compiled, not run on hardware
//
#include <hip/hip_runtime.h>
#include <hip/hip_bf16.h>

typedef __attribute__((ext_vector_type(16))) _Float16 v16h;
typedef __attribute__((ext_vector_type(8)))  _Float16 v8h;
typedef __attribute__((ext_vector_type(8)))  float    v8f;
typedef __attribute__((ext_vector_type(4)))  unsigned int u32x4;
typedef __attribute__((ext_vector_type(8)))  int          i32x8;
typedef __attribute__((ext_vector_type(4)))  int          i32x4;

#define EMBED 1024
#define SEQ   2048
#define NH    16
#define HD    64
#define NBATCH 2
#define MTOT  (NBATCH * SEQ)   // 4096 rows for the big GEMMs

#if defined(__has_builtin)
#  if __has_builtin(__builtin_amdgcn_tensor_load_to_lds)
#    define USE_TDM 1
#  endif
#endif
#ifndef USE_TDM
#  define USE_TDM 0
#endif

// ---------------------------------------------------------------------------
// WMMA f16 operand loaders; k order is this file's own (A and B match), not the ISA table order.
// A (16x32, MxK): lane L holds row M=L%16; 16 contiguous K at (L<16?0:16).
// B (32x16, KxN): lane L holds col N=L%16; 16 contiguous K at (L<16?0:16).
// C/D (16x16 f32): VGPR r: lanes 0-15 -> (M=r, N=lane); 16-31 -> (M=r+8,N=lane-16)
// ---------------------------------------------------------------------------

static __device__ __forceinline__ v16h wmmaA_from_f32(const float* rowPtr, int kb, int lane) {
  const int koff = (lane & 16) ? 16 : 0;
  const float4* p0 = reinterpret_cast<const float4*>(rowPtr + kb + koff);
  const float4* p1 = reinterpret_cast<const float4*>(rowPtr + kb + 8 + koff);
  float4 a = p0[0], b = p0[1], c = p1[0], d = p1[1];
  v16h r;
  r[0]  = (_Float16)a.x; r[1]  = (_Float16)a.y; r[2]  = (_Float16)a.z; r[3]  = (_Float16)a.w;
  r[4]  = (_Float16)b.x; r[5]  = (_Float16)b.y; r[6]  = (_Float16)b.z; r[7]  = (_Float16)b.w;
  r[8]  = (_Float16)c.x; r[9]  = (_Float16)c.y; r[10] = (_Float16)c.z; r[11] = (_Float16)c.w;
  r[12] = (_Float16)d.x; r[13] = (_Float16)d.y; r[14] = (_Float16)d.z; r[15] = (_Float16)d.w;
  return r;
}

static __device__ __forceinline__ v16h wmmaB_from_f32(const float* p) {
  const float4* q = reinterpret_cast<const float4*>(p);
  float4 a = q[0], b = q[1], c = q[2], d = q[3];
  v16h r;
  r[0]  = (_Float16)a.x; r[1]  = (_Float16)a.y; r[2]  = (_Float16)a.z; r[3]  = (_Float16)a.w;
  r[4]  = (_Float16)b.x; r[5]  = (_Float16)b.y; r[6]  = (_Float16)b.z; r[7]  = (_Float16)b.w;
  r[8]  = (_Float16)c.x; r[9]  = (_Float16)c.y; r[10] = (_Float16)c.z; r[11] = (_Float16)c.w;
  r[12] = (_Float16)d.x; r[13] = (_Float16)d.y; r[14] = (_Float16)d.z; r[15] = (_Float16)d.w;
  return r;
}

static __device__ __forceinline__ v16h wmmaA_from_f16(const _Float16* rowPtr, int kb, int lane) {
  const int koff = (lane & 16) ? 16 : 0;
  v8h lo = *reinterpret_cast<const v8h*>(rowPtr + kb + koff);
  v8h hi = *reinterpret_cast<const v8h*>(rowPtr + kb + 8 + koff);
  v16h r;
#pragma unroll
  for (int i = 0; i < 8; ++i) { r[i] = lo[i]; r[8 + i] = hi[i]; }
  return r;
}

static __device__ __forceinline__ v16h wmmaB_from_f16(const _Float16* p) {
  return *reinterpret_cast<const v16h*>(p);
}

static __device__ __forceinline__ v8f wmma_f16(v16h a, v16h b, v8f c) {
  return __builtin_amdgcn_wmma_f32_16x16x32_f16(false, a, false, b, (short)0, c, false, false);
}

#if USE_TDM
// ---------------------------------------------------------------------------
// Tensor Data Mover: 2-D tile load (f16 elements) global -> LDS.
// D# built per CDNA5 ISA 8.3/8.4: group0 = {count=1, lds_addr, global_addr,
// type=2}; group1 = {mask=0, data_size=2B, tensor_dim0/1, tile_dim0/1,
// tensor_dim0_stride}; groups 2/3 zero (<=2-D tile, unused dims = 0).
// Issued by one wave; completion via S_WAIT_TENSORCNT, visibility via barrier.
// ---------------------------------------------------------------------------
static __device__ __forceinline__ void tdm_load_tile_f16(
    unsigned int lds_off, const void* gptr,
    unsigned int tile_d0, unsigned int tile_d1,
    unsigned int tensor_d0, unsigned int tensor_d1,
    unsigned long long d0_stride_elems) {
  unsigned long long ga = (unsigned long long)(uintptr_t)gptr;
  u32x4 g0;
  g0[0] = 1u;                                   // count=1 (valid user descriptor)
  g0[1] = lds_off;                              // LDS byte address
  g0[2] = (unsigned int)ga;                     // global_addr[31:0]
  g0[3] = (unsigned int)((ga >> 32) & 0x1FFFFFFu) | (2u << 30);  // addr[56:32] | type=2
  i32x8 g1;
  g1[0] = (int)(1u << 16);                      // wg_mask=0, data_size=1 (2 bytes)
  g1[1] = (int)((tensor_d0 & 0xFFFFu) << 16);   // abar_addr=0, tensor_dim0[15:0]
  g1[2] = (int)((tensor_d0 >> 16) | ((tensor_d1 & 0xFFFFu) << 16));
  g1[3] = (int)((tensor_d1 >> 16) | (tile_d0 << 16));
  g1[4] = (int)(tile_d1 & 0xFFFFu);             // tile_dim1 (tile_dim2 = 0)
  g1[5] = (int)(unsigned int)d0_stride_elems;   // tensor_dim0_stride[31:0]
  g1[6] = (int)((unsigned int)(d0_stride_elems >> 32) & 0xFFFFu);  // stride[47:32]
  g1[7] = 0;
  i32x4 z4 = {0, 0, 0, 0};
#if __clang_major__ >= 23
  i32x8 z8 = {0, 0, 0, 0, 0, 0, 0, 0};
  __builtin_amdgcn_tensor_load_to_lds(g0, g1, z4, z4, z8, 0);
#else
  __builtin_amdgcn_tensor_load_to_lds(g0, g1, z4, z4, 0);
#endif
}
#endif  // USE_TDM

// ---------------------------------------------------------------------------
// Kernel 1: projection GEMM  y = x @ W^T + bias, emitted as f16 in the layout
// the attention kernel wants.  MODE 0: Q (B,H,S,D), scaled by 1/sqrt(D).
// MODE 1: K (B,H,S,D).  MODE 2: V transposed (B,H,D,S).
// Block: 128 threads = 4 waves; wave computes a 16(M)x64(N) strip; block 64x64.
// ---------------------------------------------------------------------------
template <int MODE>
__global__ __launch_bounds__(128) void proj_kernel(const float* __restrict__ X,
                                                   const float* __restrict__ W,
                                                   const float* __restrict__ bias,
                                                   _Float16* __restrict__ out) {
  const int lane  = threadIdx.x & 31;
  const int wave  = threadIdx.x >> 5;
  const int Mbase = blockIdx.x * 64 + wave * 16;
  const int Nbase = blockIdx.y * 64;

  v8f acc[4] = {};
  const float* xrow = X + (size_t)(Mbase + (lane & 15)) * EMBED;
  const int bko = (lane & 16) ? 16 : 0;

  for (int kb = 0; kb < EMBED; kb += 32) {
    v16h a = wmmaA_from_f32(xrow, kb, lane);
#pragma unroll
    for (int j = 0; j < 4; ++j) {
      const float* wp = W + (size_t)(Nbase + j * 16 + (lane & 15)) * EMBED + kb + bko;
      v16h bm = wmmaB_from_f32(wp);
      acc[j] = wmma_f16(a, bm, acc[j]);
    }
  }

#pragma unroll
  for (int j = 0; j < 4; ++j) {
    const int n  = Nbase + j * 16 + (lane & 15);
    const float bval = bias[n];
    const int h = n >> 6, d = n & 63;
#pragma unroll
    for (int r = 0; r < 8; ++r) {
      const int m = Mbase + r + ((lane & 16) ? 8 : 0);
      const int b = m >> 11, s = m & 2047;
      float val = acc[j][r] + bval;
      if (MODE == 0) {
        out[(((size_t)(b * NH + h)) * SEQ + s) * HD + d] = (_Float16)(val * 0.125f);
      } else if (MODE == 1) {
        out[(((size_t)(b * NH + h)) * SEQ + s) * HD + d] = (_Float16)val;
      } else {
        out[(((size_t)(b * NH + h)) * HD + d) * SEQ + s] = (_Float16)val;
      }
    }
  }
}

// ---------------------------------------------------------------------------
// Kernel 2: flash attention per (b,h).  Block = 4 waves, each wave owns 16
// query rows (block covers 64).  Per 64-key block: K/V tiles are DMA'd once
// into LDS by the Tensor Data Mover (wave 0 issues tensor_load_to_lds, waits
// TENSORcnt, block barrier) instead of every wave re-loading them from global.
// 8 WMMA for QK^T, online softmax via lane shuffles in the C layout, P staged
// through LDS (C-layout -> A-layout), 8 WMMA for P@V (V pre-transposed so the
// B operand is K-contiguous).  Per-head "quantum" rescale is the closed form
// cos(qw/S + qb) because each softmax row sums to exactly 1 => mean == 1/S.
// ---------------------------------------------------------------------------
__global__ __launch_bounds__(128) void flash_kernel(const _Float16* __restrict__ Q,
                                                    const _Float16* __restrict__ K,
                                                    const _Float16* __restrict__ Vt,
                                                    const float* __restrict__ qw,
                                                    const float* __restrict__ qb,
                                                    _Float16* __restrict__ O) {
  __shared__ __align__(32) _Float16 plds[4][16][64];
#if USE_TDM
  __shared__ __align__(32) _Float16 ktile[64][64];   // [t-local][d]
  __shared__ __align__(32) _Float16 vtile[64][64];   // [d][t-local]
#endif

  const int lane = threadIdx.x & 31;
  const int wave = threadIdx.x >> 5;
  const int h = blockIdx.y;
  const int b = blockIdx.z;
  const int q0 = blockIdx.x * 64 + wave * 16;

  const size_t headOff = ((size_t)(b * NH + h)) * SEQ * HD;
  const _Float16* Qh = Q  + headOff;
  const _Float16* Kh = K  + headOff;
  const _Float16* Vh = Vt + headOff;   // (HD, SEQ) layout

  float hscale = 1.0f;
  if (h < 2) hscale = __cosf(qw[h] * (1.0f / (float)SEQ) + qb[h]);

  // Query tile (A operand), K-dim = 64 split into two 32-chunks; resident all loop.
  const _Float16* qrow = Qh + (size_t)(q0 + (lane & 15)) * HD;
  v16h aq0 = wmmaA_from_f16(qrow, 0, lane);
  v16h aq1 = wmmaA_from_f16(qrow, 32, lane);

  v8f oacc[4] = {};
  float mrow[8], lrow[8];
#pragma unroll
  for (int r = 0; r < 8; ++r) { mrow[r] = -3.0e38f; lrow[r] = 0.0f; }

  const int bko = (lane & 16) ? 16 : 0;
  const int rowHalf = (lane & 16) ? 8 : 0;

  for (int tb = 0; tb < SEQ; tb += 64) {
#if USE_TDM
    // ---- one wave DMAs the K and V tiles for this key block into LDS ----
    if (wave == 0) {
      tdm_load_tile_f16((unsigned int)(uintptr_t)&ktile[0][0],
                        Kh + (size_t)tb * HD,
                        /*tile*/ 64, 64, /*tensor*/ 64, 1u << 20,
                        /*stride*/ HD);
      tdm_load_tile_f16((unsigned int)(uintptr_t)&vtile[0][0],
                        Vh + tb,
                        /*tile*/ 64, 64, /*tensor*/ SEQ, 1u << 20,
                        /*stride*/ SEQ);
      __builtin_amdgcn_s_wait_tensorcnt(0);
    }
    __syncthreads();  // tiles visible to all 4 waves
#endif

    // ---- scores S = (Q/sqrt(D)) @ K^T for a 16x64 strip ----
    v8f sacc[4] = {};
#pragma unroll
    for (int j = 0; j < 4; ++j) {
#if USE_TDM
      const _Float16* kp = &ktile[j * 16 + (lane & 15)][bko];
#else
      const _Float16* kp = Kh + (size_t)(tb + j * 16 + (lane & 15)) * HD + bko;
#endif
      v16h b0 = wmmaB_from_f16(kp);
      v16h b1 = wmmaB_from_f16(kp + 32);
      sacc[j] = wmma_f16(aq0, b0, sacc[j]);
      sacc[j] = wmma_f16(aq1, b1, sacc[j]);
    }

    // ---- online softmax: row stats live per half-wave (16 lanes) ----
    float corr[8];
#pragma unroll
    for (int r = 0; r < 8; ++r) {
      float mx = fmaxf(fmaxf(sacc[0][r], sacc[1][r]), fmaxf(sacc[2][r], sacc[3][r]));
#pragma unroll
      for (int d = 8; d >= 1; d >>= 1) mx = fmaxf(mx, __shfl_xor(mx, d, 32));
      float mn = fmaxf(mrow[r], mx);
      corr[r] = __expf(mrow[r] - mn);
      mrow[r] = mn;
    }

    float rs[8];
#pragma unroll
    for (int r = 0; r < 8; ++r) rs[r] = 0.0f;
#pragma unroll
    for (int j = 0; j < 4; ++j) {
#pragma unroll
      for (int r = 0; r < 8; ++r) {
        float pv = __expf(sacc[j][r] - mrow[r]);
        rs[r] += pv;
        plds[wave][r + rowHalf][j * 16 + (lane & 15)] = (_Float16)pv;
      }
    }
#pragma unroll
    for (int r = 0; r < 8; ++r) {
      float t = rs[r];
#pragma unroll
      for (int d = 8; d >= 1; d >>= 1) t += __shfl_xor(t, d, 32);
      lrow[r] = lrow[r] * corr[r] + t;
#pragma unroll
      for (int j = 0; j < 4; ++j) oacc[j][r] *= corr[r];
    }

    __syncthreads();  // P tile visible (C layout -> LDS), waits DScnt

    // ---- O += P @ V  (P re-read from LDS in A layout; V^T gives K-contig B) ----
    const _Float16* prow = &plds[wave][lane & 15][0];
    v16h pa0 = wmmaA_from_f16(prow, 0, lane);
    v16h pa1 = wmmaA_from_f16(prow, 32, lane);
#pragma unroll
    for (int j = 0; j < 4; ++j) {
#if USE_TDM
      const _Float16* vp = &vtile[j * 16 + (lane & 15)][bko];
#else
      const _Float16* vp = Vh + (size_t)(j * 16 + (lane & 15)) * SEQ + tb + bko;
#endif
      v16h b0 = wmmaB_from_f16(vp);
      v16h b1 = wmmaB_from_f16(vp + 32);
      oacc[j] = wmma_f16(pa0, b0, oacc[j]);
      oacc[j] = wmma_f16(pa1, b1, oacc[j]);
    }

    __syncthreads();  // all waves done with K/V/P tiles before next DMA/overwrite
  }

  // ---- normalize, apply per-head quantum scale, emit O as (B,S,E) f16 ----
#pragma unroll
  for (int j = 0; j < 4; ++j) {
#pragma unroll
    for (int r = 0; r < 8; ++r) {
      const int s = q0 + r + rowHalf;
      float val = oacc[j][r] / lrow[r] * hscale;
      O[((size_t)(b * SEQ + s)) * EMBED + h * HD + j * 16 + (lane & 15)] = (_Float16)val;
    }
  }
}

// ---------------------------------------------------------------------------
// Kernel 3: output projection  out = O @ Wo^T + bo  (f16 A, f32 B->f16, f32 C)
// ---------------------------------------------------------------------------
__global__ __launch_bounds__(128) void oproj_kernel(const _Float16* __restrict__ A,
                                                    const float* __restrict__ W,
                                                    const float* __restrict__ bias,
                                                    float* __restrict__ out) {
  const int lane  = threadIdx.x & 31;
  const int wave  = threadIdx.x >> 5;
  const int Mbase = blockIdx.x * 64 + wave * 16;
  const int Nbase = blockIdx.y * 64;

  v8f acc[4] = {};
  const _Float16* arow = A + (size_t)(Mbase + (lane & 15)) * EMBED;
  const int bko = (lane & 16) ? 16 : 0;

  for (int kb = 0; kb < EMBED; kb += 32) {
    v16h a = wmmaA_from_f16(arow, kb, lane);
#pragma unroll
    for (int j = 0; j < 4; ++j) {
      const float* wp = W + (size_t)(Nbase + j * 16 + (lane & 15)) * EMBED + kb + bko;
      v16h bm = wmmaB_from_f32(wp);
      acc[j] = wmma_f16(a, bm, acc[j]);
    }
  }

#pragma unroll
  for (int j = 0; j < 4; ++j) {
    const int n = Nbase + j * 16 + (lane & 15);
    const float bval = bias[n];
#pragma unroll
    for (int r = 0; r < 8; ++r) {
      const int m = Mbase + r + ((lane & 16) ? 8 : 0);
      out[(size_t)m * EMBED + n] = acc[j][r] + bval;
    }
  }
}

// ---------------------------------------------------------------------------
extern "C" void kernel_launch(void* const* d_in, const int* in_sizes, int n_in,
                              void* d_out, int out_size, void* d_ws, size_t ws_size,
                              hipStream_t stream) {
  (void)in_sizes; (void)n_in; (void)out_size; (void)ws_size;

  const float* x  = (const float*)d_in[0];
  const float* Wq = (const float*)d_in[1];
  const float* bq = (const float*)d_in[2];
  const float* Wk = (const float*)d_in[3];
  const float* bk = (const float*)d_in[4];
  const float* Wv = (const float*)d_in[5];
  const float* bv = (const float*)d_in[6];
  const float* Wo = (const float*)d_in[7];
  const float* bo = (const float*)d_in[8];
  const float* qw = (const float*)d_in[9];
  const float* qb = (const float*)d_in[10];
  float* out = (float*)d_out;

  const size_t elems = (size_t)NBATCH * SEQ * EMBED;  // 4,194,304 f16 per buffer
  _Float16* Qb  = (_Float16*)d_ws;
  _Float16* Kb  = Qb + elems;
  _Float16* Vtb = Kb + elems;
  _Float16* Ob  = Vtb + elems;

  dim3 blk(128);
  dim3 gproj(MTOT / 64, EMBED / 64);
  proj_kernel<0><<<gproj, blk, 0, stream>>>(x, Wq, bq, Qb);
  proj_kernel<1><<<gproj, blk, 0, stream>>>(x, Wk, bk, Kb);
  proj_kernel<2><<<gproj, blk, 0, stream>>>(x, Wv, bv, Vtb);

  dim3 gfa(SEQ / 64, NH, NBATCH);
  flash_kernel<<<gfa, blk, 0, stream>>>(Qb, Kb, Vtb, qw, qb, Ob);

  oproj_kernel<<<gproj, blk, 0, stream>>>(Ob, Wo, bo, out);
}
